// IDRIS_MODEL_37357625540781
// MI455X (gfx1250) — matrix operand order repaired and checked by reading, not compiled, not run on hardware
//
#include <hip/hip_runtime.h>
#include <stdint.h>

#define TSTEPS 288
#define BATCH  4096
#define NIN    8
#define NH     50
#define ROWS   32            // batch rows per block
#define MT     2             // 16-row M tiles
#define NWAVES 13            // 13 N-tiles cover 208 interleaved gate columns
#define NTHREADS (NWAVES * 32)
#define KAUG   64            // augmented K: [h(50) | x(8) | 1 | pad(5)]
#define ZSTR   72            // LDS row stride (halves): 144B -> conflict-free b128 loads
#define NGCOL  208           // 52 units * 4 gates (units 50,51 are zero pad)

typedef __attribute__((ext_vector_type(16))) __bf16       v16bf;
typedef __attribute__((ext_vector_type(8)))  float        v8f;
typedef __attribute__((ext_vector_type(4)))  unsigned int u32x4;
typedef __attribute__((ext_vector_type(8)))  int          i32x8;
typedef __attribute__((ext_vector_type(4)))  int          i32x4;

union FragBF { v16bf v; int4 q[2]; };

__device__ __forceinline__ unsigned short f2bf(float f) {
  unsigned u = __float_as_uint(f);
  return (unsigned short)((u + 0x7FFFu + ((u >> 16) & 1u)) >> 16);  // RNE
}

// fast sigmoid: v_exp_f32 + v_rcp_f32 (no IEEE div chain)
__device__ __forceinline__ float fast_sigmoid(float v) {
  return __builtin_amdgcn_rcpf(1.f + __expf(-v));
}

__device__ __forceinline__ float fast_tanh(float v) {
#if __has_builtin(__builtin_amdgcn_tanhf)
  return __builtin_amdgcn_tanhf(v);                  // gfx1250 v_tanh_f32
#else
  float e2 = __expf(-2.f * v);
  return (1.f - e2) * __builtin_amdgcn_rcpf(1.f + e2);
#endif
}

// TDM: 2D strided tile (tile_dim0=8 fp32 contiguous, tile_dim1=rows,
// tensor_dim0_stride = T*8 elements) from global x into LDS.
__device__ __forceinline__ void tdm_load_x_tile(const float* gsrc, void* ldst, int rows) {
  unsigned long long ga = (unsigned long long)(uintptr_t)gsrc;
  unsigned lds = (unsigned)(uintptr_t)ldst;          // flat->LDS offset (addr[31:0])
  u32x4 g0;
  g0[0] = 1u;                                        // count = 1 valid descriptor
  g0[1] = lds;                                       // lds_addr
  g0[2] = (unsigned)ga;                              // global_addr lo
  g0[3] = ((unsigned)(ga >> 32) & 0x01FFFFFFu) | 0x80000000u;  // addr hi | type=2
  i32x8 g1;
  g1[0] = 0x00020000;                                // data_size = 4B, mask=0
  g1[1] = (int)(8u << 16);                           // tensor_dim0 = 8 (lo16)
  g1[2] = (int)0xFFFF0000u;                          // dim0 hi=0 | tensor_dim1 lo16
  g1[3] = (int)(8u << 16);                           // dim1 hi=0 | tile_dim0 = 8
  g1[4] = rows;                                      // tile_dim1 = rows, tile_dim2 = 0
  g1[5] = TSTEPS * NIN;                              // tensor_dim0_stride = 2304 elems
  g1[6] = 0;
  g1[7] = 0;
  i32x4 z4 = {0, 0, 0, 0};
#if __clang_major__ >= 23
  i32x8 z8 = {0, 0, 0, 0, 0, 0, 0, 0};
  __builtin_amdgcn_tensor_load_to_lds(g0, g1, z4, z4, z8, 0);
#else
  __builtin_amdgcn_tensor_load_to_lds(g0, g1, z4, z4, 0);
#endif
}

__global__ __launch_bounds__(NTHREADS)
void IDRIS_lstm_fused(const float* __restrict__ x,    const float* __restrict__ W_ih,
                      const float* __restrict__ W_hh, const float* __restrict__ b_ih,
                      const float* __restrict__ b_hh, const float* __restrict__ W1,
                      const float* __restrict__ b1,   const float* __restrict__ W2,
                      const float* __restrict__ b2,   const float* __restrict__ W3,
                      const float* __restrict__ b3,   float* __restrict__ out) {
  __shared__ __align__(16) unsigned short z_lds[ROWS * ZSTR];   // augmented A rows (bf16)
  __shared__ __align__(16) union WU {
    unsigned short w[NGCOL * ZSTR];                             // interleaved weights (bf16)
    float fc[NGCOL * ZSTR / 2];                                 // reused by FC head
  } wu;
  __shared__ __align__(16) float xbuf[2][ROWS * NIN];           // TDM double buffer

  const int tid  = threadIdx.x;
  const int lane = tid & 31;
  const int wave = tid >> 5;
  const int gsel = lane & 3;           // gate within interleave [i,f,g,o]
  const int jloc = (lane >> 2) & 3;    // unit within N-tile
  const int hi   = (lane >> 4) & 1;
  const int unit = wave * 4 + jloc;    // hidden unit 0..51
  const int bRow0 = blockIdx.x * ROWS;

  // ---- build LDS: zero h, constant-1 column 58; bf16 gate-interleaved weights ----
  for (int idx = tid; idx < ROWS * ZSTR; idx += NTHREADS) {
    int k = idx % ZSTR;
    z_lds[idx] = (k == 58) ? (unsigned short)0x3F80 : (unsigned short)0;
  }
  for (int idx = tid; idx < NGCOL * ZSTR; idx += NTHREADS) {
    int n = idx / ZSTR, k = idx % ZSTR;
    int j = n >> 2, g = n & 3;
    float val = 0.f;
    if (j < NH && k < KAUG) {
      int no = g * NH + j;                           // torch gate order [i,f,g,o]
      if (k < NH)             val = W_hh[no * NH + k];
      else if (k < NH + NIN)  val = W_ih[no * NIN + (k - NH)];
      else if (k == 58)       val = b_ih[no] + b_hh[no];
    }
    wu.w[idx] = f2bf(val);
  }

  // ---- prologue: TDM x_0, x_1 ----
  if (wave == 0) {
    tdm_load_x_tile(x + (size_t)bRow0 * TSTEPS * NIN, &xbuf[0][0], ROWS);
    tdm_load_x_tile(x + (size_t)bRow0 * TSTEPS * NIN + NIN, &xbuf[1][0], ROWS);
    __builtin_amdgcn_s_wait_tensorcnt(1);            // x_0 complete
  }
  __syncthreads();

  for (int e = tid; e < ROWS * NIN; e += NTHREADS)   // stage x_0 into A rows
    z_lds[(e >> 3) * ZSTR + NH + (e & 7)] = f2bf(xbuf[0][e]);

  // ---- preload B fragments (weights) into VGPRs for the whole recurrence ----
  FragBF bfrag[2];
  {
    const int nn = wave * 16 + (lane & 15);
#pragma unroll
    for (int kc = 0; kc < 2; ++kc) {
      const int base = kc * 32 + ((lane & 16) ? 16 : 0);
      const int4* p = reinterpret_cast<const int4*>(&wu.w[nn * ZSTR + base]);
      bfrag[kc].q[0] = p[0];
      bfrag[kc].q[1] = p[1];
    }
  }
  __syncthreads();

  float c_reg[MT][8], h_reg[MT][8];
#pragma unroll
  for (int mt = 0; mt < MT; ++mt)
#pragma unroll
    for (int r = 0; r < 8; ++r) { c_reg[mt][r] = 0.f; h_reg[mt][r] = 0.f; }

  for (int t = 0; t < TSTEPS; ++t) {
    // -------- issue ALL A-fragment loads first (overlap LDS latency) --------
    FragBF afrag[MT][2];
#pragma unroll
    for (int mt = 0; mt < MT; ++mt) {
      const int arow = mt * 16 + (lane & 15);
      const int ab   = (lane & 16) ? 16 : 0;
#pragma unroll
      for (int kc = 0; kc < 2; ++kc) {
        afrag[mt][kc].q[0] =
            *reinterpret_cast<const int4*>(&z_lds[arow * ZSTR + kc * 32 + ab]);
        afrag[mt][kc].q[1] =
            *reinterpret_cast<const int4*>(&z_lds[arow * ZSTR + kc * 32 + ab + 8]);
      }
    }

    // -------- gates = [h|x|1] @ Wt^T via WMMA; activations; c/h update --------
#pragma unroll
    for (int mt = 0; mt < MT; ++mt) {
      v8f acc = {};
      acc = __builtin_amdgcn_wmma_f32_16x16x32_bf16(false, afrag[mt][0].v, false,
                                                    bfrag[0].v, (short)0, acc,
                                                    false, false);
      acc = __builtin_amdgcn_wmma_f32_16x16x32_bf16(false, afrag[mt][1].v, false,
                                                    bfrag[1].v, (short)0, acc,
                                                    false, false);
#pragma unroll
      for (int r = 0; r < 8; ++r) {
        float v  = acc[r];
        // i,f,o lanes: sigmoid(v); g lane: tanh(v) = 2*sigmoid(2v)-1
        float vs = (gsel == 2) ? 2.f * v : v;
        float s  = fast_sigmoid(vs);
        float a  = (gsel == 2) ? (2.f * s - 1.f) : s;
        float fg = __shfl_down(a, 1, 32);            // gather f,g,o into the i-lane
        float gg = __shfl_down(a, 2, 32);
        float og = __shfl_down(a, 3, 32);
        float cn = fg * c_reg[mt][r] + a * gg;       // meaningful in gsel==0 lanes
        c_reg[mt][r] = cn;
        h_reg[mt][r] = og * fast_tanh(cn);
      }
    }

    // -------- overlap: DMA x_{t+2}; guarantee x_{t+1} landed --------
    if (wave == 0) {
      if (t + 2 < TSTEPS) {
        tdm_load_x_tile(x + (size_t)bRow0 * TSTEPS * NIN + (size_t)(t + 2) * NIN,
                        &xbuf[t & 1][0], ROWS);
        __builtin_amdgcn_s_wait_tensorcnt(1);        // in-order: x_{t+1} done
      } else {
        __builtin_amdgcn_s_wait_tensorcnt(0);
      }
    }
    __syncthreads();                                 // z_lds reads done; x_{t+1} visible

    // -------- publish h_t (bf16) and stage x_{t+1} --------
    if (gsel == 0 && unit < NH) {
#pragma unroll
      for (int mt = 0; mt < MT; ++mt)
#pragma unroll
        for (int r = 0; r < 8; ++r) {
          int row = mt * 16 + r + 8 * hi;
          z_lds[row * ZSTR + unit] = f2bf(h_reg[mt][r]);
          if (t == TSTEPS - 1) wu.fc[row * 52 + unit] = h_reg[mt][r];  // f32 final h
        }
    }
    if (t + 1 < TSTEPS) {
      for (int e = tid; e < ROWS * NIN; e += NTHREADS)
        z_lds[(e >> 3) * ZSTR + NH + (e & 7)] = f2bf(xbuf[(t + 1) & 1][e]);
    }
    __syncthreads();
  }

  // -------- FC head: 50 -> 30 -> 30 -> 6 (no nonlinearities) --------
  float* hbuf = wu.fc;                 // [ROWS][52]
  float* ybuf = wu.fc + ROWS * 52;     // [ROWS][30]
  for (int p = tid; p < ROWS * 30; p += NTHREADS) {
    int row = p / 30, o = p % 30;
    float acc = b1[o];
    for (int k = 0; k < NH; ++k) acc += hbuf[row * 52 + k] * W1[o * NH + k];
    ybuf[p] = acc;
  }
  __syncthreads();
  float t2[(ROWS * 30 + NTHREADS - 1) / NTHREADS];
  int np = 0;
  for (int p = tid; p < ROWS * 30; p += NTHREADS, ++np) {
    int row = p / 30, o = p % 30;
    float acc = b2[o];
    for (int k = 0; k < 30; ++k) acc += ybuf[row * 30 + k] * W2[o * 30 + k];
    t2[np] = acc;
  }
  __syncthreads();
  np = 0;
  for (int p = tid; p < ROWS * 30; p += NTHREADS, ++np) ybuf[p] = t2[np];
  __syncthreads();
  for (int p = tid; p < ROWS * 6; p += NTHREADS) {
    int row = p / 6, o = p % 6;
    float acc = b3[o];
    for (int k = 0; k < 30; ++k) acc += ybuf[row * 30 + k] * W3[o * 30 + k];
    out[(size_t)(bRow0 + row) * 6 + o] = acc;
  }
}

extern "C" void kernel_launch(void* const* d_in, const int* in_sizes, int n_in,
                              void* d_out, int out_size, void* d_ws, size_t ws_size,
                              hipStream_t stream) {
  (void)in_sizes; (void)n_in; (void)out_size; (void)d_ws; (void)ws_size;
  const float* x    = (const float*)d_in[0];
  const float* W_ih = (const float*)d_in[1];
  const float* W_hh = (const float*)d_in[2];
  const float* b_ih = (const float*)d_in[3];
  const float* b_hh = (const float*)d_in[4];
  const float* W1   = (const float*)d_in[5];
  const float* b1   = (const float*)d_in[6];
  const float* W2   = (const float*)d_in[7];
  const float* b2   = (const float*)d_in[8];
  const float* W3   = (const float*)d_in[9];
  const float* b3   = (const float*)d_in[10];
  dim3 grid(BATCH / ROWS), block(NTHREADS);
  IDRIS_lstm_fused<<<grid, block, 0, stream>>>(x, W_ih, W_hh, b_ih, b_hh,
                                               W1, b1, W2, b2, W3, b3,
                                               (float*)d_out);
}
